// PointNet2_34574486733453
// MI455X (gfx1250) — compile-verified, matrix operand order checked by reading, not run on hardware
//
#include <hip/hip_runtime.h>
#include <hip/hip_bf16.h>
#include <cstdint>
#include <cstddef>

// ---------------------------------------------------------------------------
// PointNet++ forward for MI455X (gfx1250).
//  * GEMMs (the FLOP-dominant part) run on v_wmma_f32_16x16x32_f16, wave32.
//    Wave tile 32x32 (4 WMMAs / K-step, 2 A + 2 B fragments -> 1 global b128
//    per WMMA), block tile 64 rows x 128 cols (8 waves, 2x4), K step 32.
//  * Activations live point-major [col][Cpad] (fp16): B-fragment = one v16h
//    global load, D store = b128s. All activations fit in the 192MB L2.
//  * Weight tiles are staged to LDS by the Tensor Data Mover (TDM) with a
//    hand-built D# descriptor + s_wait_tensorcnt (fallback: cooperative copy).
// ---------------------------------------------------------------------------

typedef __attribute__((ext_vector_type(16))) _Float16 v16h;
typedef __attribute__((ext_vector_type(8)))  _Float16 v8h;
typedef __attribute__((ext_vector_type(8)))  float    v8f;
typedef __attribute__((ext_vector_type(4)))  unsigned int u32x4;
typedef __attribute__((ext_vector_type(8)))  int i32x8;
typedef __attribute__((ext_vector_type(4)))  int i32x4;

#if defined(__HIP_DEVICE_COMPILE__) && __has_builtin(__builtin_amdgcn_tensor_load_to_lds) && __has_builtin(__builtin_amdgcn_s_wait_tensorcnt)
#define PN2_HAVE_TDM 1
#else
#define PN2_HAVE_TDM 0
#endif

#if PN2_HAVE_TDM
// Build a 2-D Tensor DMA descriptor (CDNA5 ISA) and issue
// TENSOR_LOAD_TO_LDS. data_size=2 bytes, tile = tile0 x tile1 elements.
__device__ __forceinline__ void pn2_tdm_load_2d(uint32_t lds_addr, const void* gptr,
                                                unsigned td0, unsigned td1,
                                                unsigned tile0, unsigned tile1,
                                                unsigned stride0)
{
  uint64_t ga = (uint64_t)(uintptr_t)gptr;
  u32x4 g0;
  g0[0] = 1u;                                          // count=1, user descriptor
  g0[1] = lds_addr;                                    // lds_addr [63:32]
  g0[2] = (uint32_t)ga;                                // global_addr lo
  g0[3] = (uint32_t)((ga >> 32) & 0x1FFFFFFu) | (2u << 30);  // addr hi | type=2
  i32x8 g1;
  g1[0] = (int)(1u << 16);                             // data_size=1 -> 2 bytes
  g1[1] = (int)((td0 & 0xFFFFu) << 16);                // tensor_dim0[15:0]
  g1[2] = (int)((td0 >> 16) | ((td1 & 0xFFFFu) << 16));// dim0 hi | dim1 lo
  g1[3] = (int)((td1 >> 16) | (tile0 << 16));          // dim1 hi | tile_dim0
  g1[4] = (int)(tile1 & 0xFFFFu);                      // tile_dim1 (tile_dim2=0)
  g1[5] = (int)stride0;                                // tensor_dim0_stride lo
  g1[6] = 0;                                           // stride hi | dim1_stride lo
  g1[7] = 0;
  i32x4 z4 = {0, 0, 0, 0};
#if defined(__clang_major__) && (__clang_major__ >= 23)
  i32x8 z8 = {0, 0, 0, 0, 0, 0, 0, 0};
  __builtin_amdgcn_tensor_load_to_lds(g0, g1, z4, z4, z8, 0);
#else
  __builtin_amdgcn_tensor_load_to_lds(g0, g1, z4, z4, 0);
#endif
}
#endif

// ---------------------------------------------------------------------------
// GEMM: Y[col][COp] = relu(Wp[COp][Kp] * X[col][XS rows used: Kp] + bp)
// Block: 256 threads = 8 waves; block tile 64 rows x 128 cols.
// Wave (wm,wn): rows [co0+wm*32, +32), cols [n0+wn*32, +32): 4 WMMAs / K-step.
// ---------------------------------------------------------------------------
__global__ __launch_bounds__(256) void pn2_gemm_wmma(
    const _Float16* __restrict__ Wp, const float* __restrict__ bp,
    const _Float16* __restrict__ X, int XS,
    _Float16* __restrict__ Y, int COp, int Kp, int relu)
{
  __shared__ __attribute__((aligned(128))) _Float16 lds_w[64 * 32];
  const int lane  = threadIdx.x & 31;
  const int wave  = threadIdx.x >> 5;
  const int wm    = wave >> 2;            // 0..1 : 32-row slice
  const int wn    = wave & 3;             // 0..3 : 32-col slice
  const int rlane = lane & 15;
  const int khalf = lane >> 4;
  const int ksel  = khalf << 4;
  const long n0   = (long)blockIdx.x * 128;
  const int  co0  = blockIdx.y * 64;
  const long colA = n0 + (long)(wn * 32 + rlane);

  v8f acc00 = {}, acc01 = {}, acc10 = {}, acc11 = {};
  for (int k = 0; k < Kp; k += 32) {
    __syncthreads();                       // protect lds_w against prior readers
#if PN2_HAVE_TDM
    if (wave == 0) {
      pn2_tdm_load_2d((uint32_t)(uintptr_t)lds_w,
                      Wp + (size_t)co0 * Kp + k,
                      (unsigned)Kp, (unsigned)COp, 32u, 64u, (unsigned)Kp);
      __builtin_amdgcn_s_wait_tensorcnt(0);
    }
#else
    for (int i = threadIdx.x; i < 64 * 32; i += 256)
      lds_w[i] = Wp[(size_t)(co0 + (i >> 5)) * Kp + k + (i & 31)];
#endif
    __syncthreads();

    const _Float16* lw = &lds_w[((wm << 5) + rlane) * 32 + ksel];
    v16h a0 = *(const v16h*)lw;
    v16h a1 = *(const v16h*)(lw + 16 * 32);
    const _Float16* xb = X + colA * (long)XS + k + ksel;
    v16h b0 = *(const v16h*)xb;
    v16h b1 = *(const v16h*)(xb + (long)16 * XS);
    __builtin_prefetch(xb + 32, 0, 1);     // global_prefetch next K slab
    acc00 = __builtin_amdgcn_wmma_f32_16x16x32_f16(false, a0, false, b0,
                                                   (short)0, acc00, false, false);
    acc01 = __builtin_amdgcn_wmma_f32_16x16x32_f16(false, a0, false, b1,
                                                   (short)0, acc01, false, false);
    acc10 = __builtin_amdgcn_wmma_f32_16x16x32_f16(false, a1, false, b0,
                                                   (short)0, acc10, false, false);
    acc11 = __builtin_amdgcn_wmma_f32_16x16x32_f16(false, a1, false, b1,
                                                   (short)0, acc11, false, false);
  }

  const int r0 = co0 + (wm << 5) + (khalf << 3);   // 8 consecutive out rows
  v8h o00, o01, o10, o11;
#pragma unroll
  for (int r = 0; r < 8; ++r) {
    float bb0 = bp[r0 + r];
    float bb1 = bp[r0 + 16 + r];
    float v00 = acc00[r] + bb0, v01 = acc01[r] + bb0;
    float v10 = acc10[r] + bb1, v11 = acc11[r] + bb1;
    if (relu) {
      v00 = fmaxf(v00, 0.f); v01 = fmaxf(v01, 0.f);
      v10 = fmaxf(v10, 0.f); v11 = fmaxf(v11, 0.f);
    }
    o00[r] = (_Float16)v00; o01[r] = (_Float16)v01;
    o10[r] = (_Float16)v10; o11[r] = (_Float16)v11;
  }
  _Float16* y0 = Y + colA * (long)COp + r0;        // column colA
  *(v8h*)y0 = o00;
  *(v8h*)(y0 + 16) = o10;
  _Float16* y1 = y0 + (long)16 * COp;              // column colA+16
  *(v8h*)y1 = o01;
  *(v8h*)(y1 + 16) = o11;
}

// --------------------------- weight packing (f32 -> padded f16) -------------
__global__ void pn2_pack_w(const float* __restrict__ W, _Float16* __restrict__ Wp,
                           int CO, int CI, int CIp, int total)
{
  int t = blockIdx.x * blockDim.x + threadIdx.x;
  if (t >= total) return;
  int r = t / CIp, c = t % CIp;
  Wp[t] = (r < CO && c < CI) ? (_Float16)W[r * CI + c] : (_Float16)0.0f;
}
__global__ void pn2_pack_b(const float* __restrict__ b, float* __restrict__ bp,
                           int CO, int COp)
{
  int t = blockIdx.x * blockDim.x + threadIdx.x;
  if (t >= COp) return;
  bp[t] = (t < CO) ? b[t] : 0.f;
}

// --------------------------- farthest point sampling ------------------------
// One block per batch; wave32 shuffle argmax + one small LDS stage per step.
__global__ void pn2_fps(const float* __restrict__ xyz, float* __restrict__ mind,
                        int* __restrict__ fidx, int Np, int npoint)
{
  const int b = blockIdx.x;
  const float* P = xyz + (size_t)b * Np * 3;
  float* md = mind + (size_t)b * Np;
  const int lane = threadIdx.x & 31;
  const int wid  = threadIdx.x >> 5;
  __shared__ float swv[8];
  __shared__ int   swi[8];
  __shared__ int   s_last;
  for (int i = threadIdx.x; i < Np; i += 256) md[i] = 1e10f;
  if (threadIdx.x == 0) { fidx[b * npoint] = 0; s_last = 0; }
  __syncthreads();
  for (int s = 1; s < npoint; ++s) {
    int last = s_last;
    float lx = P[last * 3 + 0], ly = P[last * 3 + 1], lz = P[last * 3 + 2];
    float best = -1.f; int bi = 0;
    for (int i = threadIdx.x; i < Np; i += 256) {
      float dx = P[i * 3 + 0] - lx, dy = P[i * 3 + 1] - ly, dz = P[i * 3 + 2] - lz;
      float m = fminf(md[i], dx * dx + dy * dy + dz * dz);
      md[i] = m;
      if (m > best) { best = m; bi = i; }
    }
    // wave32 argmax reduction
    for (int off = 16; off > 0; off >>= 1) {
      float ov = __shfl_down(best, off);
      int   oi = __shfl_down(bi, off);
      if (ov > best) { best = ov; bi = oi; }
    }
    if (lane == 0) { swv[wid] = best; swi[wid] = bi; }
    __syncthreads();
    if (threadIdx.x == 0) {
      float bb = swv[0]; int ii = swi[0];
      for (int w = 1; w < 8; ++w)
        if (swv[w] > bb) { bb = swv[w]; ii = swi[w]; }
      s_last = ii; fidx[b * npoint + s] = ii;
    }
    __syncthreads();
  }
}

__global__ void pn2_gather_xyz(const float* __restrict__ xyz,
                               const int* __restrict__ fidx,
                               float* __restrict__ nx, int Np, int S, int total)
{
  int t = blockIdx.x * blockDim.x + threadIdx.x;
  if (t >= total) return;
  int b = t / S;
  const float* p = xyz + ((size_t)b * Np + fidx[t]) * 3;
  nx[t * 3 + 0] = p[0]; nx[t * 3 + 1] = p[1]; nx[t * 3 + 2] = p[2];
}

// --------------------------- ball query (one wave32 per center) -------------
__global__ void pn2_ball_query(const float* __restrict__ xyz,
                               const float* __restrict__ nxyz,
                               int* __restrict__ out, int Np, int S,
                               float r2, int ns, int nwaves)
{
  int gw = (blockIdx.x * blockDim.x + threadIdx.x) >> 5;
  int lane = threadIdx.x & 31;
  if (gw >= nwaves) return;                  // uniform per wave
  int b = gw / S, s = gw % S;
  float cx = nxyz[(b * S + s) * 3 + 0];
  float cy = nxyz[(b * S + s) * 3 + 1];
  float cz = nxyz[(b * S + s) * 3 + 2];
  const float* P = xyz + (size_t)b * Np * 3;
  int* o = out + (size_t)(b * S + s) * ns;
  int cnt = 0, first = 0;
  for (int base = 0; base < Np && cnt < ns; base += 32) {
    int i = base + lane;
    bool in = false;
    if (i < Np) {
      float dx = P[i * 3 + 0] - cx, dy = P[i * 3 + 1] - cy, dz = P[i * 3 + 2] - cz;
      in = (dx * dx + dy * dy + dz * dz) < r2;
    }
    unsigned m = (unsigned)__ballot(in);     // wave32: low 32 bits valid
    while (m && cnt < ns) {
      int t = __builtin_ctz(m); m &= m - 1;  // in index order (ref top-k trick)
      int pi = base + t;
      if (cnt == 0) first = pi;
      if (lane == 0) o[cnt] = pi;
      ++cnt;
    }
  }
  if (lane == 0) for (int j = cnt; j < ns; ++j) o[j] = (cnt == 0) ? 0 : first;
}

// --------------------------- SA grouping (build point-major X) --------------
__global__ void pn2_sa_group(const float* __restrict__ xyz,
                             const float* __restrict__ nxyz,
                             const int* __restrict__ idx,
                             const _Float16* __restrict__ feat, int C, int Cp,
                             _Float16* __restrict__ X, int CIp,
                             int Np, int S, int ns, long total)
{
  long t = (long)blockIdx.x * blockDim.x + threadIdx.x;
  if (t >= total) return;
  long bs = t / ns;
  int  b  = (int)(bs / S);
  int  pi = idx[t];
  const float* p = xyz + ((size_t)b * Np + pi) * 3;
  const float* c = nxyz + bs * 3;
  _Float16* xo = X + t * CIp;
  xo[0] = (_Float16)(p[0] - c[0]);
  xo[1] = (_Float16)(p[1] - c[1]);
  xo[2] = (_Float16)(p[2] - c[2]);
  if (feat) {
    const _Float16* f = feat + ((size_t)b * Np + pi) * Cp;
    for (int ch = 0; ch < C; ++ch) xo[3 + ch] = f[ch];
  } else {
    for (int ch = 0; ch < C; ++ch) xo[3 + ch] = (_Float16)p[ch];
  }
  for (int ch = 3 + C; ch < CIp; ++ch) xo[ch] = (_Float16)0.0f;
}

// --------------------------- max-pool over nsample (point-major) ------------
__global__ void pn2_maxpool(const _Float16* __restrict__ Y, _Float16* __restrict__ out,
                            int Cp, int ns, long total)
{
  long t = (long)blockIdx.x * blockDim.x + threadIdx.x;
  if (t >= total) return;
  long col = t / Cp; int c = (int)(t % Cp);
  float m = -3.0e38f;
  for (int j = 0; j < ns; ++j)
    m = fmaxf(m, (float)Y[(col * ns + j) * (long)Cp + c]);
  out[t] = (_Float16)m;
}

// --------------------------- FP: 3-NN interpolate + concat ------------------
__global__ void pn2_fp_build(const float* __restrict__ pts1, const float* __restrict__ pts2,
                             const _Float16* __restrict__ fea2, int C2, int C2p,
                             const _Float16* __restrict__ fea1, int C1, int C1p,
                             _Float16* __restrict__ X, int CIp,
                             int n1, int n2, long total)
{
  long t = (long)blockIdx.x * blockDim.x + threadIdx.x;
  if (t >= total) return;
  int b = (int)(t / n1);
  float px = pts1[t * 3 + 0], py = pts1[t * 3 + 1], pz = pts1[t * 3 + 2];
  const float* Q = pts2 + (size_t)b * n2 * 3;
  float d0 = 3.4e38f, d1 = 3.4e38f, d2 = 3.4e38f;
  int j0 = 0, j1 = 0, j2 = 0;
  for (int j = 0; j < n2; ++j) {
    float dx = Q[j * 3 + 0] - px, dy = Q[j * 3 + 1] - py, dz = Q[j * 3 + 2] - pz;
    float dd = dx * dx + dy * dy + dz * dz;
    if (dd < d0)      { d2 = d1; j2 = j1; d1 = d0; j1 = j0; d0 = dd; j0 = j; }
    else if (dd < d1) { d2 = d1; j2 = j1; d1 = dd; j1 = j; }
    else if (dd < d2) { d2 = dd; j2 = j; }
  }
  float w0 = 1.f / (d0 + 1e-8f), w1 = 1.f / (d1 + 1e-8f), w2 = 1.f / (d2 + 1e-8f);
  float ws = w0 + w1 + w2;
  w0 /= ws; w1 /= ws; w2 /= ws;
  const _Float16* f0 = fea2 + ((size_t)b * n2 + j0) * C2p;
  const _Float16* f1 = fea2 + ((size_t)b * n2 + j1) * C2p;
  const _Float16* f2 = fea2 + ((size_t)b * n2 + j2) * C2p;
  _Float16* xo = X + t * CIp;
  for (int c = 0; c < C2; ++c)
    xo[c] = (_Float16)(w0 * (float)f0[c] + w1 * (float)f1[c] + w2 * (float)f2[c]);
  if (fea1) {
    const _Float16* g = fea1 + (size_t)t * C1p;
    for (int c = 0; c < C1; ++c) xo[C2 + c] = g[c];
  }
  for (int c = C2 + C1; c < CIp; ++c) xo[c] = (_Float16)0.0f;
}

// --------------------------- final heads ------------------------------------
__global__ void pn2_heads(const _Float16* __restrict__ l0f, const _Float16* __restrict__ xf,
                          const float* __restrict__ Wo, const float* __restrict__ bo,
                          const float* __restrict__ Wc, const float* __restrict__ bc,
                          float* __restrict__ pf, float* __restrict__ score,
                          float* __restrict__ off, long BN)
{
  long n = (long)blockIdx.x * blockDim.x + threadIdx.x;
  if (n >= BN) return;
  const _Float16* l = l0f + n * 128;
  const _Float16* x = xf + n * 128;
  for (int c = 0; c < 128; ++c) pf[n * 128 + c] = (float)l[c];
  float s = bc[0];
  for (int c = 0; c < 128; ++c) s += Wc[c] * (float)x[c];
  score[n] = 1.f / (1.f + __expf(-s));
  for (int o = 0; o < 3; ++o) {
    float v = bo[o];
    for (int c = 0; c < 128; ++c) v += Wo[o * 128 + c] * (float)x[c];
    off[n * 3 + o] = v;
  }
}

// ===========================================================================
// Host orchestration
// ===========================================================================
extern "C" void kernel_launch(void* const* d_in, const int* in_sizes, int n_in,
                              void* d_out, int out_size, void* d_ws, size_t ws_size,
                              hipStream_t stream)
{
  const int Bb = 8, Nn = 16384;

  // ---- map inputs: handle both insertion-order and sorted-pytree order ----
  bool points_first = (in_sizes[0] == Bb * Nn * 3);
  const float* points = (const float*)(points_first ? d_in[0] : d_in[n_in - 1]);
  // canonical group ids: 0..3 sa1..sa4, 4 fp4, 5 fp3, 6 fp2, 7 fp1, 8 shared,
  // 9 offset, 10 cls
  const int gcnt[11] = {3, 3, 3, 3, 2, 2, 2, 3, 1, 1, 1};
  int start[11];
  if (points_first) {
    int idx = 1;
    for (int g = 0; g < 11; ++g) { start[g] = idx; idx += gcnt[g] * 2; }
  } else {
    const int sorted_ord[11] = {10, 7, 6, 5, 4, 9, 0, 1, 2, 3, 8};
    int idx = 0;
    for (int s = 0; s < 11; ++s) { int g = sorted_ord[s]; start[g] = idx; idx += gcnt[g] * 2; }
  }
  auto Wof = [&](int g, int l) { return (const float*)d_in[start[g] + 2 * l]; };
  auto Bof = [&](int g, int l) { return (const float*)d_in[start[g] + 2 * l + 1]; };

  // ---- workspace bump allocator ----
  uintptr_t cur = (uintptr_t)d_ws;
  auto alloc = [&](size_t bytes) {
    uintptr_t p = (cur + 255) & ~(uintptr_t)255;
    cur = p + bytes;
    return (void*)p;
  };

  // ---- layers (canonical order) ----
  struct LS { int g, l, co, ci; };
  const LS layers[22] = {
    {0,0,32,6},{0,1,32,32},{0,2,64,32},
    {1,0,64,67},{1,1,64,64},{1,2,128,64},
    {2,0,128,131},{2,1,128,128},{2,2,256,128},
    {3,0,256,259},{3,1,256,256},{3,2,512,256},
    {4,0,256,768},{4,1,256,256},
    {5,0,256,384},{5,1,256,256},
    {6,0,256,320},{6,1,128,256},
    {7,0,128,128},{7,1,128,128},{7,2,128,128},
    {8,0,128,128},
  };
  _Float16* Wp[22]; float* Bp[22]; int COp[22], KP[22];
  for (int i = 0; i < 22; ++i) {
    int cop = (layers[i].co + 63) & ~63;
    int cip = (layers[i].ci + 31) & ~31;
    COp[i] = cop; KP[i] = cip;
    Wp[i] = (_Float16*)alloc((size_t)cop * cip * sizeof(_Float16));
    Bp[i] = (float*)alloc((size_t)cop * sizeof(float));
    int total = cop * cip;
    pn2_pack_w<<<(total + 255) / 256, 256, 0, stream>>>(
        Wof(layers[i].g, layers[i].l), Wp[i], layers[i].co, layers[i].ci, cip, total);
    pn2_pack_b<<<(cop + 255) / 256, 256, 0, stream>>>(
        Bof(layers[i].g, layers[i].l), Bp[i], layers[i].co, cop);
  }

  // ---- scratch buffers ----
  float* mind = (float*)alloc((size_t)Bb * Nn * sizeof(float));
  int*   fidx1 = (int*)alloc((size_t)Bb * 256 * 4);
  int*   fidx2 = (int*)alloc((size_t)Bb * 128 * 4);
  int*   fidx3 = (int*)alloc((size_t)Bb * 64 * 4);
  int*   fidx4 = (int*)alloc((size_t)Bb * 16 * 4);
  float* nx1 = (float*)alloc((size_t)Bb * 256 * 3 * 4);
  float* nx2 = (float*)alloc((size_t)Bb * 128 * 3 * 4);
  float* nx3 = (float*)alloc((size_t)Bb * 64 * 3 * 4);
  float* nx4 = (float*)alloc((size_t)Bb * 16 * 3 * 4);
  int*   bi1 = (int*)alloc((size_t)Bb * 256 * 16 * 4);
  int*   bi2 = (int*)alloc((size_t)Bb * 128 * 16 * 4);
  int*   bi3 = (int*)alloc((size_t)Bb * 64 * 16 * 4);
  int*   bi4 = (int*)alloc((size_t)Bb * 16 * 16 * 4);
  const size_t BIG = (size_t)Bb * Nn * 128;            // 16.78M f16 elems
  _Float16* bufA = (_Float16*)alloc(BIG * sizeof(_Float16));
  _Float16* bufB = (_Float16*)alloc(BIG * sizeof(_Float16));
  _Float16* l0f  = (_Float16*)alloc(BIG * sizeof(_Float16));
  _Float16* l1f  = (_Float16*)alloc((size_t)Bb * 256 * 64 * sizeof(_Float16));
  _Float16* l2f  = (_Float16*)alloc((size_t)Bb * 128 * 128 * sizeof(_Float16));
  _Float16* l3f  = (_Float16*)alloc((size_t)Bb * 64 * 256 * sizeof(_Float16));
  _Float16* l4f  = (_Float16*)alloc((size_t)Bb * 16 * 512 * sizeof(_Float16));
  _Float16* l3fb = (_Float16*)alloc((size_t)Bb * 64 * 256 * sizeof(_Float16));
  _Float16* l2fb = (_Float16*)alloc((size_t)Bb * 128 * 256 * sizeof(_Float16));
  _Float16* l1fb = (_Float16*)alloc((size_t)Bb * 256 * 128 * sizeof(_Float16));
  (void)ws_size; (void)out_size;

  auto gemm = [&](int li, const _Float16* Xp, int XS, _Float16* Yp, long NC) {
    pn2_gemm_wmma<<<dim3((unsigned)(NC / 128), (unsigned)(COp[li] / 64)), 256, 0, stream>>>(
        Wp[li], Bp[li], Xp, XS, Yp, COp[li], KP[li], 1);
  };

  // ---------------- SA1: 16384 -> 256 centers ----------------
  pn2_fps<<<Bb, 256, 0, stream>>>(points, mind, fidx1, Nn, 256);
  pn2_gather_xyz<<<(Bb * 256 + 255) / 256, 256, 0, stream>>>(points, fidx1, nx1, Nn, 256, Bb * 256);
  pn2_ball_query<<<(Bb * 256 + 7) / 8, 256, 0, stream>>>(points, nx1, bi1, Nn, 256, 0.01f, 16, Bb * 256);
  {
    long tot = (long)Bb * 256 * 16;
    pn2_sa_group<<<(unsigned)((tot + 255) / 256), 256, 0, stream>>>(
        points, nx1, bi1, (const _Float16*)nullptr, 3, 0, bufA, 32, Nn, 256, 16, tot);
  }
  gemm(0, bufA, 32, bufB, 32768);
  gemm(1, bufB, 64, bufA, 32768);
  gemm(2, bufA, 64, bufB, 32768);
  pn2_maxpool<<<((Bb * 256 * 64) + 255) / 256, 256, 0, stream>>>(bufB, l1f, 64, 16, (long)Bb * 256 * 64);

  // ---------------- SA2: 256 -> 128 ----------------
  pn2_fps<<<Bb, 256, 0, stream>>>(nx1, mind, fidx2, 256, 128);
  pn2_gather_xyz<<<(Bb * 128 + 255) / 256, 256, 0, stream>>>(nx1, fidx2, nx2, 256, 128, Bb * 128);
  pn2_ball_query<<<(Bb * 128 + 7) / 8, 256, 0, stream>>>(nx1, nx2, bi2, 256, 128, 0.01f, 16, Bb * 128);
  {
    long tot = (long)Bb * 128 * 16;
    pn2_sa_group<<<(unsigned)((tot + 255) / 256), 256, 0, stream>>>(
        nx1, nx2, bi2, l1f, 64, 64, bufA, 96, 256, 128, 16, tot);
  }
  gemm(3, bufA, 96, bufB, 16384);
  gemm(4, bufB, 64, bufA, 16384);
  gemm(5, bufA, 64, bufB, 16384);
  pn2_maxpool<<<((Bb * 128 * 128) + 255) / 256, 256, 0, stream>>>(bufB, l2f, 128, 16, (long)Bb * 128 * 128);

  // ---------------- SA3: 128 -> 64 ----------------
  pn2_fps<<<Bb, 256, 0, stream>>>(nx2, mind, fidx3, 128, 64);
  pn2_gather_xyz<<<(Bb * 64 + 255) / 256, 256, 0, stream>>>(nx2, fidx3, nx3, 128, 64, Bb * 64);
  pn2_ball_query<<<(Bb * 64 + 7) / 8, 256, 0, stream>>>(nx2, nx3, bi3, 128, 64, 0.04f, 16, Bb * 64);
  {
    long tot = (long)Bb * 64 * 16;
    pn2_sa_group<<<(unsigned)((tot + 255) / 256), 256, 0, stream>>>(
        nx2, nx3, bi3, l2f, 128, 128, bufA, 160, 128, 64, 16, tot);
  }
  gemm(6, bufA, 160, bufB, 8192);
  gemm(7, bufB, 128, bufA, 8192);
  gemm(8, bufA, 128, bufB, 8192);
  pn2_maxpool<<<((Bb * 64 * 256) + 255) / 256, 256, 0, stream>>>(bufB, l3f, 256, 16, (long)Bb * 64 * 256);

  // ---------------- SA4: 64 -> 16 ----------------
  pn2_fps<<<Bb, 256, 0, stream>>>(nx3, mind, fidx4, 64, 16);
  pn2_gather_xyz<<<(Bb * 16 + 255) / 256, 256, 0, stream>>>(nx3, fidx4, nx4, 64, 16, Bb * 16);
  pn2_ball_query<<<(Bb * 16 + 7) / 8, 256, 0, stream>>>(nx3, nx4, bi4, 64, 16, 0.16f, 16, Bb * 16);
  {
    long tot = (long)Bb * 16 * 16;
    pn2_sa_group<<<(unsigned)((tot + 255) / 256), 256, 0, stream>>>(
        nx3, nx4, bi4, l3f, 256, 256, bufA, 288, 64, 16, 16, tot);
  }
  gemm(9, bufA, 288, bufB, 2048);
  gemm(10, bufB, 256, bufA, 2048);
  gemm(11, bufA, 256, bufB, 2048);
  pn2_maxpool<<<((Bb * 16 * 512) + 255) / 256, 256, 0, stream>>>(bufB, l4f, 512, 16, (long)Bb * 16 * 512);

  // ---------------- FP4: interp l4 -> l3 pts ----------------
  pn2_fp_build<<<(Bb * 64 + 255) / 256, 256, 0, stream>>>(
      nx3, nx4, l4f, 512, 512, l3f, 256, 256, bufA, 768, 64, 16, (long)Bb * 64);
  gemm(12, bufA, 768, bufB, 512);
  gemm(13, bufB, 256, l3fb, 512);

  // ---------------- FP3 ----------------
  pn2_fp_build<<<(Bb * 128 + 255) / 256, 256, 0, stream>>>(
      nx2, nx3, l3fb, 256, 256, l2f, 128, 128, bufA, 384, 128, 64, (long)Bb * 128);
  gemm(14, bufA, 384, bufB, 1024);
  gemm(15, bufB, 256, l2fb, 1024);

  // ---------------- FP2 ----------------
  pn2_fp_build<<<(Bb * 256 + 255) / 256, 256, 0, stream>>>(
      nx1, nx2, l2fb, 256, 256, l1f, 64, 64, bufA, 320, 256, 128, (long)Bb * 256);
  gemm(16, bufA, 320, bufB, 2048);
  gemm(17, bufB, 256, l1fb, 2048);

  // ---------------- FP1: interp to all 16384 points ----------------
  pn2_fp_build<<<((Bb * Nn) + 255) / 256, 256, 0, stream>>>(
      points, nx1, l1fb, 128, 128, (const _Float16*)nullptr, 0, 0,
      bufA, 128, Nn, 256, (long)Bb * Nn);
  gemm(18, bufA, 128, bufB, (long)Bb * Nn);
  gemm(19, bufB, 128, bufA, (long)Bb * Nn);
  gemm(20, bufA, 128, l0f, (long)Bb * Nn);

  // ---------------- shared MLP + heads ----------------
  gemm(21, l0f, 128, bufA, (long)Bb * Nn);
  {
    float* out  = (float*)d_out;
    float* pf   = out;
    float* sc   = out + (size_t)Bb * Nn * 128;
    float* offp = sc + (size_t)Bb * Nn;
    pn2_heads<<<((Bb * Nn) + 255) / 256, 256, 0, stream>>>(
        l0f, bufA, Wof(9, 0), Bof(9, 0), Wof(10, 0), Bof(10, 0),
        pf, sc, offp, (long)Bb * Nn);
  }
}
